// CSPNet_48936857370784
// MI455X (gfx1250) — compile-verified, matrix operand order checked by reading, not run on hardware
//
#include <hip/hip_runtime.h>
#include <hip/hip_bf16.h>

typedef __bf16 bf16_t;
typedef __bf16  v16bf __attribute__((ext_vector_type(16)));
typedef float   v8f   __attribute__((ext_vector_type(8)));

#define GG    512
#define AA    24
#define NN    (GG*AA)
#define HID   128
#define TDIM  256
#define NFREQ 10
#define EPG   (AA*AA)      /* 576 edges per graph          */
#define ETILE 32
#define NET   (EPG/ETILE)  /* 18 edge tiles per graph      */
#define K1P   352          /* EDGE_IN 322 padded to 11*32  */
#define KS1   11           /* K-steps for GEMM1            */
#define KS2   4            /* K-steps for GEMM2            */
#define KLP   416          /* LAT_IN 400 padded to 13*32   */
#define NL    4

/* bf16 workspace layout (element offsets) */
#define OFF_EMB 0
#define SZ_EMB  (128*64)
#define OFF_LAT (OFF_EMB + SZ_EMB)
#define SZ_LAT  (128*KLP)
#define OFF_WE1 (OFF_LAT + SZ_LAT)
#define SZ_WE1  (128*K1P)
#define OFF_WE2 (OFF_WE1 + 4*SZ_WE1)
#define SZ_WE2  (128*128)
#define OFF_WN1 (OFF_WE2 + 4*SZ_WE2)
#define SZ_WN1  (128*256)
#define OFF_WN2 (OFF_WN1 + 4*SZ_WN1)
#define SZ_WN2  (128*128)

__device__ __forceinline__ v8f splat8(float x) {
    v8f v = {x, x, x, x, x, x, x, x};
    return v;
}

__device__ __forceinline__ float silu(float x) {
    return x / (1.f + __expf(-x));
}

__device__ __forceinline__ float fracf1(float x) { return x - floorf(x); }

/* Load a 16x32 bf16 WMMA fragment from a row-major [rows][pitch] matrix.
 * Lane L<16 : row L,     K halves {k..k+7, k+16..k+23}
 * Lane L>=16: row L-16,  K halves {k+8..k+15, k+24..k+31}            */
__device__ __forceinline__ v16bf frag_ld(const bf16_t* tile, int pitch, int lane) {
    const bf16_t* p = tile + (lane & 15) * pitch + ((lane >> 4) << 3);
    union { v16bf v; uint4 q[2]; } u;
    u.q[0] = *reinterpret_cast<const uint4*>(p);
    u.q[1] = *reinterpret_cast<const uint4*>(p + 16);
    return u.v;
}

__device__ __forceinline__ v8f wmma_bf16(v16bf a, v16bf b, v8f c) {
    return __builtin_amdgcn_wmma_f32_16x16x32_bf16(false, a, false, b, (short)0, c,
                                                   false, false);
}

/* acc[m] += A[m-tile] x B-strip ; A row-major [..][apitch], B = W^T [128][bpitch] */
template <int MT>
__device__ __forceinline__ void gemm_accum(const bf16_t* A, int apitch, int kbaseA,
                                           const bf16_t* B, int bpitch, int kbaseB,
                                           int ksteps, int n0, int lane, v8f* acc) {
    for (int ks = 0; ks < ksteps; ++ks) {
        v16bf bf = frag_ld(B + n0 * bpitch + kbaseB + ks * 32, bpitch, lane);
#pragma unroll
        for (int m = 0; m < MT; ++m) {
            v16bf af = frag_ld(A + (m * 16) * apitch + kbaseA + ks * 32, apitch, lane);
            acc[m] = wmma_bf16(af, bf, acc[m]);
        }
    }
}

/* ---------------- weight transpose + pad + bf16 convert --------------------- */
__global__ void cvt_wT_kernel(const float* __restrict__ src, bf16_t* __restrict__ dst,
                              int IN, int OUT, int KP) {
    int idx = blockIdx.x * blockDim.x + threadIdx.x;
    int total = OUT * KP;
    if (idx >= total) return;
    int o = idx / KP, k = idx - o * KP;
    float v = (k < IN) ? src[k * OUT + o] : 0.f;
    dst[idx] = (bf16_t)v;
}

/* ---------------------------- per-graph megakernel -------------------------- */
__global__ __launch_bounds__(256, 1) void cspnet_graph_kernel(
    const float* __restrict__ t_in, const float* __restrict__ bb,
    const float* __restrict__ fc_in, const float* __restrict__ so3,
    const float* __restrict__ lat_in, const float* __restrict__ b_emb,
    const float* __restrict__ b_lat, const float* __restrict__ be1,
    const float* __restrict__ be2, const float* __restrict__ bn1,
    const float* __restrict__ bn2, const float* __restrict__ W_coord,
    const float* __restrict__ W_lattice, const float* __restrict__ W_so3,
    const bf16_t* __restrict__ ws, float* __restrict__ out) {
    __shared__ __align__(16) float  sh_h[AA * 128];      /* 12.0 KB residual h  */
    __shared__ __align__(16) float  sh_agg[AA * 128];    /* 12.0 KB agg / gfeat */
    __shared__ __align__(16) bf16_t sh_hbf[32 * 128];    /*  8.0 KB h bf16      */
    __shared__ __align__(16) bf16_t sh_big[32 * K1P];    /* 22.0 KB ein/h0/nin  */
    __shared__ __align__(16) bf16_t sh_t1[32 * 128];     /*  8.0 KB t1          */
    __shared__ __align__(16) float  sh_fc[32 * 3];
    __shared__ __align__(16) float  sh_lat[8];

    const int g = blockIdx.x;
    const int tid = threadIdx.x;
    const int lane = tid & 31;
    const int n0 = (tid >> 5) * 16;       /* N-strip owned by this wave */
    const int rlane = (lane >> 4) << 3;   /* D-tile row base for this lane */
    const int nlane = lane & 15;          /* D-tile column for this lane   */

    /* ---- init ---- */
    for (int idx = tid; idx < 32 * 128; idx += 256) sh_hbf[idx] = (bf16_t)0.f;
    if (tid < AA * 3) sh_fc[tid] = fc_in[g * AA * 3 + tid];
    if (tid < 6) sh_lat[tid] = lat_in[g * 6 + tid];

    /* ---- prologue: bb tile [32][64] ---- */
    for (int idx = tid; idx < 32 * 64; idx += 256) {
        int r = idx >> 6, c = idx & 63;
        sh_big[idx] = (bf16_t)((r < AA) ? bb[(g * AA + r) * 64 + c] : 0.f);
    }
    __syncthreads();

    /* emb = bb @ W_emb + b_emb (no activation) */
    v8f eacc[2];
    eacc[0] = splat8(b_emb[n0 + nlane]);
    eacc[1] = eacc[0];
    gemm_accum<2>(sh_big, 64, 0, ws + OFF_EMB, 64, 0, 2, n0, lane, eacc);
    __syncthreads();

    /* h0 phase A: cols 0..223 as [32][224] : [emb | so3 | t(0..79)] */
#pragma unroll
    for (int m = 0; m < 2; ++m)
#pragma unroll
        for (int r = 0; r < 8; ++r)
            sh_big[(m * 16 + rlane + r) * 224 + n0 + nlane] = (bf16_t)eacc[m][r];
    for (int idx = tid; idx < 32 * 96; idx += 256) {
        int r = idx / 96, gc = 128 + idx % 96;
        float v = 0.f;
        if (r < AA)
            v = (gc < 144) ? so3[(g * AA + r) * 16 + gc - 128]
                           : t_in[g * TDIM + gc - 144];
        sh_big[r * 224 + gc] = (bf16_t)v;
    }
    __syncthreads();

    v8f hacc[2];
    hacc[0] = splat8(b_lat[n0 + nlane]);
    hacc[1] = hacc[0];
    gemm_accum<2>(sh_big, 224, 0, ws + OFF_LAT, KLP, 0, 7, n0, lane, hacc);
    __syncthreads();

    /* h0 phase B: cols 224..415 as [32][192] : [t(80..255) | pad] */
    for (int idx = tid; idx < 32 * 192; idx += 256) {
        int r = idx / 192, c = idx - (idx / 192) * 192, gc = 224 + c;
        float v = 0.f;
        if (r < AA && gc < 400) v = t_in[g * TDIM + gc - 144];
        sh_big[r * 192 + c] = (bf16_t)v;
    }
    __syncthreads();
    gemm_accum<2>(sh_big, 192, 0, ws + OFF_LAT, KLP, 224, 6, n0, lane, hacc);
#pragma unroll
    for (int m = 0; m < 2; ++m)
#pragma unroll
        for (int r = 0; r < 8; ++r) {
            int row = m * 16 + rlane + r;
            if (row < AA) {
                float x = hacc[m][r];
                sh_h[row * 128 + n0 + nlane] = x;
                sh_hbf[row * 128 + n0 + nlane] = (bf16_t)x;
            }
        }
    __syncthreads();

    /* ---------------------------- layers ---------------------------- */
    for (int l = 0; l < NL; ++l) {
        const bf16_t* We1t = ws + OFF_WE1 + l * SZ_WE1;
        const bf16_t* We2t = ws + OFF_WE2 + l * SZ_WE2;
        const bf16_t* Wn1t = ws + OFF_WN1 + l * SZ_WN1;
        const bf16_t* Wn2t = ws + OFF_WN2 + l * SZ_WN2;
        const float *b1 = be1 + l * 128, *b2 = be2 + l * 128;
        const float *c1 = bn1 + l * 128, *c2 = bn2 + l * 128;

        if (lane == 0) { /* warm node-phase weights into GL2 */
            __builtin_prefetch(Wn1t + (tid >> 5) * (SZ_WN1 / 8), 0, 1);
            __builtin_prefetch(Wn2t + (tid >> 5) * (SZ_WN2 / 8), 0, 1);
        }
        for (int idx = tid; idx < AA * 128; idx += 256) sh_agg[idx] = 0.f;

        /* Preload this wave's edge-phase B fragments into registers:
         * 11 (We1, K=352) + 4 (We2, K=128) frags = 120 VGPRs, resident
         * across all 18 edge tiles -> inner loop is ds_load + wmma only. */
        v16bf bw1[KS1], bw2[KS2];
#pragma unroll
        for (int ks = 0; ks < KS1; ++ks)
            bw1[ks] = frag_ld(We1t + n0 * K1P + ks * 32, K1P, lane);
#pragma unroll
        for (int ks = 0; ks < KS2; ++ks)
            bw2[ks] = frag_ld(We2t + n0 * 128 + ks * 32, 128, lane);
        const float b1v = b1[n0 + nlane];
        const float b2v = b2[n0 + nlane];
        __syncthreads();

        for (int et = 0; et < NET; ++et) {
            /* build ein tile [32][352]: [h_i | h_j | lat(6) | pe(60) | pad] */
            {
                int r = tid & 31, part = tid >> 5;
                int e = et * ETILE + r;
                int i = e / AA, j = e - i * AA;
                float d0 = fracf1(sh_fc[j * 3 + 0] - sh_fc[i * 3 + 0]);
                float d1 = fracf1(sh_fc[j * 3 + 1] - sh_fc[i * 3 + 1]);
                float d2 = fracf1(sh_fc[j * 3 + 2] - sh_fc[i * 3 + 2]);
                const bf16_t* hi = sh_hbf + i * 128;
                const bf16_t* hj = sh_hbf + j * 128;
                bf16_t* dst = sh_big + r * K1P;
                for (int c = part * 44; c < part * 44 + 44; ++c) {
                    bf16_t v;
                    if (c < 128) v = hi[c];
                    else if (c < 256) v = hj[c - 128];
                    else if (c < 262) v = (bf16_t)sh_lat[c - 256];
                    else if (c < 322) {
                        int s = c - 262;
                        int isCos = (s >= 30);
                        int ss = isCos ? s - 30 : s;
                        int ax = ss / 10, k = ss - ax * 10;
                        float d = (ax == 0) ? d0 : ((ax == 1) ? d1 : d2);
                        float ang = d * 6.28318530717958647f * (float)k;
                        v = (bf16_t)(isCos ? __cosf(ang) : __sinf(ang));
                    } else v = (bf16_t)0.f;
                    dst[c] = v;
                }
            }
            __syncthreads();
            /* GEMM1: t1 = silu(ein @ We1 + b1)  (B register-resident) */
            {
                v8f acc[2];
                acc[0] = splat8(b1v);
                acc[1] = acc[0];
#pragma unroll
                for (int ks = 0; ks < KS1; ++ks) {
#pragma unroll
                    for (int m = 0; m < 2; ++m) {
                        v16bf af = frag_ld(sh_big + (m * 16) * K1P + ks * 32, K1P, lane);
                        acc[m] = wmma_bf16(af, bw1[ks], acc[m]);
                    }
                }
#pragma unroll
                for (int m = 0; m < 2; ++m)
#pragma unroll
                    for (int r = 0; r < 8; ++r)
                        sh_t1[(m * 16 + rlane + r) * 128 + n0 + nlane] =
                            (bf16_t)silu(acc[m][r]);
            }
            __syncthreads();
            /* GEMM2: ef = silu(t1 @ We2 + b2); scatter-add to agg[i] */
            {
                v8f acc[2];
                acc[0] = splat8(b2v);
                acc[1] = acc[0];
#pragma unroll
                for (int ks = 0; ks < KS2; ++ks) {
#pragma unroll
                    for (int m = 0; m < 2; ++m) {
                        v16bf af = frag_ld(sh_t1 + (m * 16) * 128 + ks * 32, 128, lane);
                        acc[m] = wmma_bf16(af, bw2[ks], acc[m]);
                    }
                }
#pragma unroll
                for (int m = 0; m < 2; ++m)
#pragma unroll
                    for (int r = 0; r < 8; ++r) {
                        int row = m * 16 + rlane + r;
                        int i = (et * ETILE + row) / AA;
                        atomicAdd(&sh_agg[i * 128 + n0 + nlane], silu(acc[m][r]));
                    }
            }
            __syncthreads();
        }

        /* agg mean (counts == 24) -> bf16 in sh_big[0:4096], rows>=24 zeroed */
        for (int idx = tid; idx < 32 * 128; idx += 256) {
            int r = idx >> 7;
            sh_big[idx] = (bf16_t)((r < AA) ? sh_agg[idx] * (1.f / 24.f) : 0.f);
        }
        __syncthreads();

        /* GEMM3: t2 = silu([h | agg] @ Wn1 + c1) -> sh_big[4096:8192] */
        {
            v8f acc[2];
            acc[0] = splat8(c1[n0 + nlane]);
            acc[1] = acc[0];
            gemm_accum<2>(sh_hbf, 128, 0, Wn1t, 256, 0, 4, n0, lane, acc);
            gemm_accum<2>(sh_big, 128, 0, Wn1t, 256, 128, 4, n0, lane, acc);
#pragma unroll
            for (int m = 0; m < 2; ++m)
#pragma unroll
                for (int r = 0; r < 8; ++r)
                    sh_big[4096 + (m * 16 + rlane + r) * 128 + n0 + nlane] =
                        (bf16_t)silu(acc[m][r]);
        }
        __syncthreads();
        /* GEMM4: h += silu(t2 @ Wn2 + c2) */
        {
            v8f acc[2];
            acc[0] = splat8(c2[n0 + nlane]);
            acc[1] = acc[0];
            gemm_accum<2>(sh_big + 4096, 128, 0, Wn2t, 128, 0, 4, n0, lane, acc);
#pragma unroll
            for (int m = 0; m < 2; ++m)
#pragma unroll
                for (int r = 0; r < 8; ++r) {
                    int row = m * 16 + rlane + r;
                    if (row < AA) {
                        float nh = sh_h[row * 128 + n0 + nlane] + silu(acc[m][r]);
                        sh_h[row * 128 + n0 + nlane] = nh;
                        sh_hbf[row * 128 + n0 + nlane] = (bf16_t)nh;
                    }
                }
        }
        __syncthreads();
    }

    /* ---------------------------- heads ---------------------------- */
    if (tid < 128) { /* gfeat (mean over nodes) into sh_agg[0:128] */
        float s = 0.f;
        for (int r = 0; r < AA; ++r) s += sh_h[r * 128 + tid];
        sh_agg[tid] = s * (1.f / 24.f);
    }
    __syncthreads();
    if (tid < 6) { /* lattice_out */
        float acc = 0.f;
        for (int k = 0; k < 128; ++k) acc += sh_agg[k] * W_lattice[k * 6 + tid];
        out[g * 6 + tid] = acc;
    }
    if (tid >= 8 && tid < 8 + AA * 3) { /* coord_out -> mod(.+frac, 1) */
        int tt = tid - 8;
        int node = tt / 3, a = tt - node * 3;
        float acc = 0.f;
        for (int k = 0; k < 128; ++k) acc += sh_h[node * 128 + k] * W_coord[k * 3 + a];
        float x = acc + sh_fc[node * 3 + a];
        out[GG * 6 + (g * AA + node) * 3 + a] = fracf1(x);
    }
    if (tid >= 96 && tid < 96 + AA) { /* so3_out (L2-normalized) */
        int node = tid - 96;
        float v0 = 0.f, v1 = 0.f, v2 = 0.f, v3 = 0.f;
        for (int k = 0; k < 128; ++k) {
            float h = sh_h[node * 128 + k];
            const float* wr = W_so3 + k * 4;
            v0 += h * wr[0]; v1 += h * wr[1]; v2 += h * wr[2]; v3 += h * wr[3];
        }
        float inv = rsqrtf(v0 * v0 + v1 * v1 + v2 * v2 + v3 * v3);
        float* o = out + GG * 6 + NN * 3 + (g * AA + node) * 4;
        o[0] = v0 * inv; o[1] = v1 * inv; o[2] = v2 * inv; o[3] = v3 * inv;
    }
}

extern "C" void kernel_launch(void* const* d_in, const int* in_sizes, int n_in,
                              void* d_out, int out_size, void* d_ws, size_t ws_size,
                              hipStream_t stream) {
    (void)in_sizes; (void)n_in; (void)out_size; (void)ws_size;
    const float* t_in   = (const float*)d_in[0];
    const float* bb     = (const float*)d_in[1];
    const float* fc     = (const float*)d_in[2];
    const float* so3    = (const float*)d_in[3];
    const float* lat    = (const float*)d_in[4];
    const float* W_emb  = (const float*)d_in[8];
    const float* b_emb  = (const float*)d_in[9];
    const float* W_lat  = (const float*)d_in[10];
    const float* b_lat  = (const float*)d_in[11];
    const float* We1    = (const float*)d_in[12];
    const float* be1    = (const float*)d_in[13];
    const float* We2    = (const float*)d_in[14];
    const float* be2    = (const float*)d_in[15];
    const float* Wn1    = (const float*)d_in[16];
    const float* bn1    = (const float*)d_in[17];
    const float* Wn2    = (const float*)d_in[18];
    const float* bn2    = (const float*)d_in[19];
    const float* W_coord   = (const float*)d_in[20];
    const float* W_lattice = (const float*)d_in[21];
    const float* W_so3     = (const float*)d_in[22];
    bf16_t* ws = (bf16_t*)d_ws;

    auto cvt = [&](const float* s, bf16_t* d, int IN, int OUT, int KP) {
        int tot = OUT * KP;
        cvt_wT_kernel<<<(tot + 255) / 256, 256, 0, stream>>>(s, d, IN, OUT, KP);
    };
    cvt(W_emb, ws + OFF_EMB, 64, 128, 64);
    cvt(W_lat, ws + OFF_LAT, 400, 128, KLP);
    for (int l = 0; l < NL; ++l) {
        cvt(We1 + l * 322 * 128, ws + OFF_WE1 + l * SZ_WE1, 322, 128, K1P);
        cvt(We2 + l * 128 * 128, ws + OFF_WE2 + l * SZ_WE2, 128, 128, 128);
        cvt(Wn1 + l * 256 * 128, ws + OFF_WN1 + l * SZ_WN1, 256, 128, 256);
        cvt(Wn2 + l * 128 * 128, ws + OFF_WN2 + l * SZ_WN2, 128, 128, 128);
    }
    cspnet_graph_kernel<<<GG, 256, 0, stream>>>(
        t_in, bb, fc, so3, lat, b_emb, b_lat, be1, be2, bn1, bn2,
        W_coord, W_lattice, W_so3, ws, (float*)d_out);
}
